// NLMFeat_22771916604114
// MI455X (gfx1250) — matrix operand order repaired and checked by reading, not compiled, not run on hardware
//
#include <hip/hip_runtime.h>
#include <hip/hip_bf16.h>

// ---------------------------------------------------------------------------
// NLM (7x7, C=64) via CDNA5 WMMA, all operands staged in LDS.
//   phase1: corr(i,j) = <x_center[i,:], x_row_dy[j,:]>   (v_wmma_f32_16x16x32_bf16)
//   weights: band-mask * exp(-(Sc+Sn-2corr)/(C*H2)), staged to LDS, reused as A
//   phase2: Y += W(16x32) x Xband(32x64); wsum via ones-matrix WMMA
//   output: transpose through LDS, coalesced float4 stores; center fp32 row
//           prefetched via global_load_async_to_lds_b128 (asynccnt path)
// ---------------------------------------------------------------------------

typedef __bf16        bf16x16 __attribute__((ext_vector_type(16)));
typedef float         f32x8   __attribute__((ext_vector_type(8)));
typedef unsigned int  u32x4   __attribute__((ext_vector_type(4)));
typedef int           i32x4   __attribute__((ext_vector_type(4)));

union FragB16 {
    bf16x16        v;
    u32x4          q[2];
    unsigned short s[16];
};

__device__ __forceinline__ unsigned short f2bf(float f) {
    union { __bf16 h; unsigned short s; } u;
    u.h = (__bf16)f;                       // hardware f32->bf16 (RNE)
    return u.s;
}
__device__ __forceinline__ float bf2f(unsigned int us) {
    union { unsigned int u; float f; } x; x.u = us << 16;
    return x.f;
}

#define WMMA_BF16(A, B, C) \
    __builtin_amdgcn_wmma_f32_16x16x32_bf16(false, (A), false, (B), (short)0, (C), false, false)

#if defined(__has_builtin)
#if __has_builtin(__builtin_amdgcn_global_load_async_to_lds_b128) && \
    __has_builtin(__builtin_amdgcn_s_wait_asynccnt)
#define HAVE_ASYNC_LDS 1
#endif
#endif

#define ASG __attribute__((address_space(1)))
#define ASL __attribute__((address_space(3)))

// geometry
#define NB   4
#define NC   64
#define NH   128
#define NW   128
#define RR   3
#define NDY  7
#define PW   144          // 8 left pad + 128 + 8 right pad
#define KEXP 1.5625f      // 1/(C*H2) = 1/(64*0.01)
#define RESS 0.2f

// LDS layout (dynamic):
//   s_pc : bf16 [7][144][64]  pixel-major  (phase-1 A/B)      129024 B
//   s_cp : bf16 [7][64][144]  channel-major (phase-2 B)       129024 B
//   s_S  : f32  [7][144]      per-pixel sum of squares          4032 B
//   s_W  : bf16 [8][16][32]   per-wave weight staging           8192 B
//   s_xc : f32  [64][128]     center row fp32 (async copy)     32768 B
#define PC_OFF 0
#define CP_OFF 129024
#define S_OFF  258048
#define W_OFF  262080
#define XC_OFF 270272
#define SMEM_BYTES 303040

__global__ void __launch_bounds__(256)
nlm_wmma_kernel(const float* __restrict__ x, float* __restrict__ out)
{
    extern __shared__ unsigned char smem[];
    unsigned short* s_pc = (unsigned short*)(smem + PC_OFF);
    unsigned short* s_cp = (unsigned short*)(smem + CP_OFF);
    float*          s_S  = (float*)(smem + S_OFF);
    unsigned short* s_W  = (unsigned short*)(smem + W_OFF);
    float*          s_xc = (float*)(smem + XC_OFF);

    const int tid = threadIdx.x;
    const int b   = blockIdx.x >> 7;
    const int r   = blockIdx.x & 127;

    // ---- zero both bf16 layouts (implements the reference zero padding) ----
    {
        unsigned int* z32 = (unsigned int*)smem;
        for (int e = tid; e < (CP_OFF + 129024) / 4; e += 256) z32[e] = 0u;
    }
    __syncthreads();

    // ---- stage 7 rows x 64 ch x 128 w, f32 -> bf16, both layouts ----
    for (int e = tid; e < NDY * NC * NW; e += 256) {
        const int w   = e & 127;
        const int c   = (e >> 7) & 63;
        const int row = e >> 13;
        const int h   = r + row - RR;
        if (h >= 0 && h < NH) {
            const float* gp = x + ((((size_t)b * NC + c) * NH + h) * NW + w);
            __builtin_prefetch(gp + 256, 0, 0);   // global_prefetch_b8
            const unsigned short hb = f2bf(*gp);
            s_pc[(row * PW + w + 8) * NC + c]  = hb;
            s_cp[(row * NC + c) * PW + w + 8]  = hb;
        }
    }

    // ---- async copy of the fp32 center row into LDS (overlaps compute) ----
#ifdef HAVE_ASYNC_LDS
    for (int e4 = tid * 4; e4 < NC * NW; e4 += 1024) {
        const float* gp = x + ((((size_t)b * NC + (e4 >> 7)) * NH + r) * NW + (e4 & 127));
        __builtin_amdgcn_global_load_async_to_lds_b128(
            (ASG i32x4*)(unsigned long long)gp,
            (ASL i32x4*)(s_xc + e4), 0, 0);
    }
#else
    for (int e = tid; e < NC * NW; e += 256)
        s_xc[e] = x[((((size_t)b * NC + (e >> 7)) * NH + r) * NW + (e & 127))];
#endif
    __syncthreads();

    // ---- per-pixel sum of squares (bf16-consistent with corr) ----
    for (int e = tid; e < NDY * PW; e += 256) {
        const unsigned int* p32 = (const unsigned int*)(s_pc + e * NC);
        float acc = 0.0f;
#pragma unroll
        for (int c = 0; c < 32; ++c) {
            unsigned int d = p32[c];
            float lo = bf2f(d & 0xFFFFu);
            float hi = bf2f(d >> 16);
            acc += lo * lo + hi * hi;
        }
        s_S[e] = acc;
    }
    __syncthreads();

    // ---- per-wave 16-pixel tile ----
    const int wave = tid >> 5;
    const int lane = tid & 31;
    const int g    = lane >> 4;
    const int n    = lane & 15;
    const int w0   = wave * 16;

    // A (center row): lane m=n -> pixel w0+n, K = 16g+{0..15}
    FragB16 fa[2];
#pragma unroll
    for (int chunk = 0; chunk < 2; ++chunk) {
        const unsigned short* ap =
            s_pc + (size_t)(3 * PW + w0 + 8 + n) * NC + chunk * 32 + 16 * g;
        fa[chunk].q[0] = *(const u32x4*)(ap);
        fa[chunk].q[1] = *(const u32x4*)(ap + 8);
    }

    // center-pixel norms + branch-free band masks (Toeplitz: jb'-i-5 in [0,6])
    float sc[8], m0f[8], m1f[8];
#pragma unroll
    for (int v = 0; v < 8; ++v) {
        const int i = v + 8 * g;
        sc[v]  = s_S[3 * PW + w0 + 8 + i];
        const int e0 = n - i - 5;
        const int e1 = n + 11 - i;
        m0f[v] = (e0 >= 0 && e0 <= 6) ? 1.0f : 0.0f;
        m1f[v] = (e1 >= 0 && e1 <= 6) ? 1.0f : 0.0f;
    }

    FragB16 fones;
#pragma unroll
    for (int k = 0; k < 16; ++k) fones.s[k] = 0x3F80;   // bf16 1.0

    f32x8 yacc[4] = {{}, {}, {}, {}};
    f32x8 ws = {};
    unsigned short* wp = s_W + wave * 512;   // [16][32] staging

#pragma unroll
    for (int dy = 0; dy < NDY; ++dy) {
        // -- phase 1: correlation against 32-pixel band (base w0-8)
        f32x8 d0 = {}, d1 = {};
#pragma unroll
        for (int chunk = 0; chunk < 2; ++chunk) {
            const int ch0 = chunk * 32 + 16 * g;   // B lane: K = 16g..16g+15
            const unsigned short* bp0 = s_pc + (size_t)(dy * PW + w0 + n)      * NC + ch0;
            const unsigned short* bp1 = s_pc + (size_t)(dy * PW + w0 + 16 + n) * NC + ch0;
            FragB16 fb0, fb1;
            fb0.q[0] = *(const u32x4*)(bp0); fb0.q[1] = *(const u32x4*)(bp0 + 8);
            fb1.q[0] = *(const u32x4*)(bp1); fb1.q[1] = *(const u32x4*)(bp1 + 8);
            d0 = WMMA_BF16(fa[chunk].v, fb0.v, d0);
            d1 = WMMA_BF16(fa[chunk].v, fb1.v, d1);
        }

        // -- weights: unconditional exp, multiplicative mask (no exec divergence)
        const float sn0 = s_S[dy * PW + w0 + n];
        const float sn1 = s_S[dy * PW + w0 + 16 + n];
#pragma unroll
        for (int v = 0; v < 8; ++v) {
            const int i = v + 8 * g;
            const float wv0 = m0f[v] * __expf(-(sc[v] + sn0 - 2.0f * d0[v]) * KEXP);
            const float wv1 = m1f[v] * __expf(-(sc[v] + sn1 - 2.0f * d1[v]) * KEXP);
            wp[i * 32 + n]      = f2bf(wv0);
            wp[i * 32 + 16 + n] = f2bf(wv1);
        }

        // -- reload W as phase-2 A fragment (same-wave LDS is in-order)
        FragB16 fw;
        fw.q[0] = *(const u32x4*)(wp + n * 32 + 16 * g);
        fw.q[1] = *(const u32x4*)(wp + n * 32 + 8 + 16 * g);

        // wsum row-reduction via ones matrix
        ws = WMMA_BF16(fw.v, fones.v, ws);

        // -- phase 2: Y += W x X_band from channel-major LDS
#pragma unroll
        for (int chunk = 0; chunk < 4; ++chunk) {
            const unsigned short* bp =
                s_cp + (size_t)(dy * NC + chunk * 16 + n) * PW + w0 + 16 * g;
            FragB16 fx;
            fx.q[0] = *(const u32x4*)(bp);
            fx.q[1] = *(const u32x4*)(bp + 8);
            yacc[chunk] = WMMA_BF16(fw.v, fx.v, yacc[chunk]);
        }
    }

    // ---- transpose Y through LDS (reuse s_pc region) ----
    __syncthreads();                       // everyone done reading s_pc/s_cp/s_S
    float* s_out = (float*)smem;           // [128 pix][64 ch]
#pragma unroll
    for (int v = 0; v < 8; ++v) {
        const int i = v + 8 * g;
        const float inv = 1.0f / ws[v];
#pragma unroll
        for (int chunk = 0; chunk < 4; ++chunk)
            s_out[(w0 + i) * NC + chunk * 16 + n] = yacc[chunk][v] * inv;
    }
#ifdef HAVE_ASYNC_LDS
    __builtin_amdgcn_s_wait_asynccnt(0);
#endif
    __syncthreads();

    // ---- residual blend + coalesced float4 stores ----
    for (int e4 = tid * 4; e4 < NC * NW; e4 += 1024) {
        const int c = e4 >> 7;
        const int w = e4 & 127;
        const float* yp = s_out + w * NC + c;
        const float y0 = yp[0], y1 = yp[NC], y2 = yp[2 * NC], y3 = yp[3 * NC];
        const float4 xc = *(const float4*)(s_xc + e4);
        float4 o;
        o.x = xc.x + RESS * (y0 - xc.x);
        o.y = xc.y + RESS * (y1 - xc.y);
        o.z = xc.z + RESS * (y2 - xc.z);
        o.w = xc.w + RESS * (y3 - xc.w);
        *(float4*)(out + ((((size_t)b * NC + c) * NH + r) * NW + w)) = o;
    }
}

extern "C" void kernel_launch(void* const* d_in, const int* in_sizes, int n_in,
                              void* d_out, int out_size, void* d_ws, size_t ws_size,
                              hipStream_t stream)
{
    (void)in_sizes; (void)n_in; (void)d_ws; (void)ws_size; (void)out_size;
    const float* x = (const float*)d_in[0];
    float* out     = (float*)d_out;
    // one workgroup per (batch,row): 4*128 = 512 blocks, 8 wave32s each
    nlm_wmma_kernel<<<dim3(NB * NH), dim3(256), SMEM_BYTES, stream>>>(x, out);
}
